// DIN_58059367907340
// MI455X (gfx1250) — matrix operand order repaired and checked by reading, not compiled, not run on hardware
//
#include <hip/hip_runtime.h>
#include <hip/hip_bf16.h>
#include <hip/hip_fp16.h>

typedef __attribute__((ext_vector_type(16))) _Float16 v16h;
typedef __attribute__((ext_vector_type(8)))  float    v8f;
typedef __attribute__((ext_vector_type(4)))  unsigned int u32x4;

#define DIN_SEQ    200
#define DIN_SEQP   208   // 13 * 16 (M padded)
#define DIN_EMBED  64
#define DIN_HIDDEN 256
#define DIN_K      256   // 4 * EMBED
#define LDH        72    // h row stride in halves (64 + 8 pad -> conflict-free)

// ---------------------------------------------------------------------------
// Pack attn_w1 (f32 [256][256], row-major K x N) into per-lane WMMA B-fragment
// layout, f16.  Fragment f = kt*16+nt; element offset = (f*32 + lane)*16 + i,
// holding w1[kt*32 + (lane>=16?8:0) + (i&7) + (i>=8?16:0)][nt*16 + (lane&15)].
// ---------------------------------------------------------------------------
__global__ void din_pack_w1(const float* __restrict__ w1,
                            _Float16* __restrict__ wp) {
    int tid  = blockIdx.x * blockDim.x + threadIdx.x;   // 65536 threads
    int i    =  tid        & 15;
    int lane = (tid >> 4)  & 31;
    int nt   = (tid >> 9)  & 15;
    int kt   = (tid >> 13) & 7;
    int n = nt * 16 + (lane & 15);
    int k = kt * 32 + ((lane >> 4) << 3) + (i & 7) + ((i >> 3) << 4);
    wp[tid] = (_Float16)w1[k * DIN_HIDDEN + n];
}

// ---------------------------------------------------------------------------
// Main fused DIN kernel: one workgroup per batch row.
// ---------------------------------------------------------------------------
__global__ __launch_bounds__(256, 2)
void din_kernel(const int*   __restrict__ hist,   // [B, SEQ]
                const int*   __restrict__ tgt,    // [B]
                const float* __restrict__ uemb,   // [VOCAB, 64]
                const float* __restrict__ iemb,   // [VOCAB, 64]
                const _Float16* __restrict__ w1p, // packed attn_w1 fragments
                const float* __restrict__ ab1,    // [256]
                const float* __restrict__ aw2,    // [256]
                const float* __restrict__ ab2,    // [1]
                const float* __restrict__ mw1,    // [128, 256]
                const float* __restrict__ mb1,    // [256]
                const float* __restrict__ mw2,    // [256]
                const float* __restrict__ mb2,    // [1]
                float* __restrict__ out) {        // [B]
    __shared__ _Float16 sH[DIN_SEQP * LDH];       // history embeddings, f16
    __shared__ float    sPart[8 * DIN_SEQP];      // per-wave partial logits
    __shared__ float    sLogit[DIN_SEQP];         // logits -> softmax weights
    __shared__ float    sT[DIN_EMBED];            // target embedding, f32
    __shared__ float    sMin[2 * DIN_EMBED];      // mlp input
    __shared__ float    sRed[8];

    const int b    = blockIdx.x;
    const int tid  = threadIdx.x;
    const int lane = tid & 31;
    const int wave = tid >> 5;

    // ---- stage target embedding + history embeddings (f32 -> f16) ----
    if (tid < DIN_EMBED)
        sT[tid] = iemb[(size_t)tgt[b] * DIN_EMBED + tid];

    const int* hrow = hist + (size_t)b * DIN_SEQ;
    for (int idx = tid; idx < DIN_SEQ * DIN_EMBED; idx += 256) {
        int s = idx >> 6, e = idx & 63;
        sH[s * LDH + e] = (_Float16)uemb[(size_t)hrow[s] * DIN_EMBED + e];
    }
    for (int idx = tid; idx < (DIN_SEQP - DIN_SEQ) * DIN_EMBED; idx += 256) {
        int s = DIN_SEQ + (idx >> 6), e = idx & 63;
        sH[s * LDH + e] = (_Float16)0.f;          // zero pad rows
    }
    __syncthreads();

    // ---- per-wave constants ----
    const int abase = (lane >> 4) << 3;           // K offset 8 for upper half-wave
    // target-embedding A fragments for K parity 0/1 (independent of M)
    v16h tf[2];
    #pragma unroll
    for (int p = 0; p < 2; ++p)
        #pragma unroll
        for (int i = 0; i < 16; ++i) {
            int c = p * 32 + abase + (i & 7) + ((i >> 3) << 4);
            tf[p][i] = (_Float16)sT[c];
        }

    // B fragments for this wave's two N tiles, resident in registers
    const int nt0 = wave * 2;
    v16h Bf[2][8];
    #pragma unroll
    for (int kt = 0; kt < 8; ++kt) {
        Bf[0][kt] = *(const v16h*)(w1p + (((kt * 16 + nt0    ) * 32) + lane) * 16);
        Bf[1][kt] = *(const v16h*)(w1p + (((kt * 16 + nt0 + 1) * 32) + lane) * 16);
    }
    const int col0 = nt0 * 16 + (lane & 15);
    const int col1 = col0 + 16;
    const float bia0 = ab1[col0], bia1 = ab1[col1];
    const float ww0  = aw2[col0], ww1  = aw2[col1];

    // ---- attention GEMM: [208 x 256] x [256 x 32(this wave)] via WMMA ----
    for (int mt = 0; mt < 13; ++mt) {
        v8f acc0 = {}; v8f acc1 = {};
        const _Float16* mrow = sH + (mt * 16 + (lane & 15)) * LDH;
        #pragma unroll
        for (int kt = 0; kt < 8; ++kt) {
            const int region = kt >> 1;           // 0:h 1:t 2:h-t 3:h*t
            const int p      = kt & 1;
            union { v16h v; struct { u32x4 lo, hi; } q; } hf;
            const _Float16* ar = mrow + p * 32 + abase;
            hf.q.lo = *(const u32x4*)(ar);
            hf.q.hi = *(const u32x4*)(ar + 16);
            v16h a;
            if      (region == 0) a = hf.v;
            else if (region == 1) a = tf[p];
            else if (region == 2) a = hf.v - tf[p];
            else                  a = hf.v * tf[p];
            acc0 = __builtin_amdgcn_wmma_f32_16x16x32_f16(
                       false, a, false, Bf[0][kt], (short)0, acc0, false, false);
            acc1 = __builtin_amdgcn_wmma_f32_16x16x32_f16(
                       false, a, false, Bf[1][kt], (short)0, acc1, false, false);
        }
        // fuse ReLU + dot with attn_w2, reduce across the 16-lane N group
        #pragma unroll
        for (int v = 0; v < 8; ++v) {
            float psum = fmaxf(acc0[v] + bia0, 0.f) * ww0
                       + fmaxf(acc1[v] + bia1, 0.f) * ww1;
            psum += __shfl_xor(psum, 1, 32);
            psum += __shfl_xor(psum, 2, 32);
            psum += __shfl_xor(psum, 4, 32);
            psum += __shfl_xor(psum, 8, 32);
            if ((lane & 15) == 0) {
                int row = mt * 16 + ((lane >> 4) << 3) + v;
                sPart[wave * DIN_SEQP + row] = psum;   // deterministic, no atomics
            }
        }
    }
    __syncthreads();

    // ---- combine per-wave partials into logits ----
    if (tid < DIN_SEQP) {
        float l = ab2[0];
        #pragma unroll
        for (int w = 0; w < 8; ++w) l += sPart[w * DIN_SEQP + tid];
        sLogit[tid] = l;
    }
    __syncthreads();

    // ---- softmax over the 200 valid positions (wave 0) ----
    if (wave == 0) {
        float m = -1e30f;
        for (int s = lane; s < DIN_SEQ; s += 32) m = fmaxf(m, sLogit[s]);
        for (int off = 16; off; off >>= 1) m = fmaxf(m, __shfl_xor(m, off, 32));
        float sum = 0.f;
        for (int s = lane; s < DIN_SEQ; s += 32) {
            float e = __expf(sLogit[s] - m);
            sLogit[s] = e;
            sum += e;
        }
        for (int off = 16; off; off >>= 1) sum += __shfl_xor(sum, off, 32);
        float inv = 1.f / sum;
        for (int s = lane; s < DIN_SEQ; s += 32) sLogit[s] *= inv;
    }
    __syncthreads();

    // ---- user_interest = sum_s att_w[s] * h[s,:]  -> mlp_in[0:64] ----
    if (tid < DIN_EMBED) {
        float acc = 0.f;
        for (int s = 0; s < DIN_SEQ; ++s)
            acc += (float)sH[s * LDH + tid] * sLogit[s];
        sMin[tid] = acc;
        sMin[DIN_EMBED + tid] = sT[tid];
    }
    __syncthreads();

    // ---- prediction head: relu(mlp_in @ mlp_w1 + b1) @ mlp_w2 + b2 ----
    float acc = mb1[tid];
    for (int k = 0; k < 2 * DIN_EMBED; ++k)
        acc = fmaf(sMin[k], mw1[k * DIN_HIDDEN + tid], acc);
    float part = fmaxf(acc, 0.f) * mw2[tid];
    for (int off = 16; off; off >>= 1) part += __shfl_xor(part, off, 32);
    if (lane == 0) sRed[wave] = part;
    __syncthreads();
    if (tid == 0) {
        float tot = mb2[0];
        #pragma unroll
        for (int w = 0; w < 8; ++w) tot += sRed[w];
        out[b] = 1.f / (1.f + __expf(-tot));
    }
}

// ---------------------------------------------------------------------------
extern "C" void kernel_launch(void* const* d_in, const int* in_sizes, int n_in,
                              void* d_out, int out_size, void* d_ws, size_t ws_size,
                              hipStream_t stream) {
    const int*   hist = (const int*)  d_in[0];
    const int*   tgt  = (const int*)  d_in[1];
    const float* uemb = (const float*)d_in[2];
    const float* iemb = (const float*)d_in[3];
    const float* aw1  = (const float*)d_in[4];
    const float* ab1  = (const float*)d_in[5];
    const float* aw2  = (const float*)d_in[6];
    const float* ab2  = (const float*)d_in[7];
    const float* mw1  = (const float*)d_in[8];
    const float* mb1  = (const float*)d_in[9];
    const float* mw2  = (const float*)d_in[10];
    const float* mb2  = (const float*)d_in[11];
    float* out = (float*)d_out;

    _Float16* w1p = (_Float16*)d_ws;   // 8*16*32*16 halves = 128 KB

    din_pack_w1<<<256, 256, 0, stream>>>(aw1, w1p);
    din_kernel<<<4096, 256, 0, stream>>>(hist, tgt, uemb, iemb, w1p,
                                         ab1, aw2, ab2,
                                         mw1, mb1, mw2, mb2, out);
}
